// SparseMultiHeadAttention_20529943675443
// MI455X (gfx1250) — matrix operand order repaired and checked by reading, not compiled, not run on hardware
//
#include <hip/hip_runtime.h>

// ---------------------------------------------------------------------------
// Problem constants (from reference): N=8192, C=1024, H=16, B=8, S=1024, D=64
// ---------------------------------------------------------------------------
#define N_TOK   8192
#define C_DIM   1024
#define H_HEADS 16
#define B_BATCH 8
#define S_SEQ   (N_TOK / B_BATCH)   // 1024
#define D_HEAD  (C_DIM / H_HEADS)   // 64
#define QK_LD   (2 * C_DIM)         // 2048 : Q|K row-major buffer
#define VT_LD   N_TOK               // 8192 : V stored transposed [C][N]

typedef __attribute__((ext_vector_type(16))) _Float16 v16h;
typedef __attribute__((ext_vector_type(8)))  float    v8f;

__device__ __forceinline__ v8f wmma_f16(const v16h a, const v16h b, const v8f c) {
  return __builtin_amdgcn_wmma_f32_16x16x32_f16(false, a, false, b,
                                                (short)0, c, false, false);
}

// ---------------------------------------------------------------------------
// WMMA fragment layouts per CDNA5 ISA 7.12.2 (wave32):
//   A 16x32 f16 : lane row m=lane&15 ; elem e -> K = 16*(lane>>4) + e (this code's order, not the ISA table)
//   B 32x16 f16 : lane col n=lane&15 ; elem e -> K = 16*(lane>>4) + e (this code's order, not the ISA table)
//   C/D 16x16 f32: elem r -> row m = r + 8*(lane>>4), col n = lane&15
// With A row-major and B stored TRANSPOSED [N][K], every fragment is two
// contiguous 8-half runs per lane -> 2x global_load_b128.
// ---------------------------------------------------------------------------
__device__ __forceinline__ v16h load_a_frag(const _Float16* __restrict__ base,
                                            int ld, int lane) {
  v16h a;
  const int m  = lane & 15;
  const int hb = (lane >> 4) << 3;
#pragma unroll
  for (int e = 0; e < 16; ++e) {
    const int k = (hb << 1) + e;
    a[e] = base[(long)m * ld + k];
  }
  return a;
}

__device__ __forceinline__ v16h load_bt_frag(const _Float16* __restrict__ baseT,
                                             int ldT, int lane) {
  v16h b;
  const int n   = lane & 15;
  const int khi = (lane >> 4) << 4;
#pragma unroll
  for (int e = 0; e < 16; ++e) {
    b[e] = baseT[(long)n * ldT + khi + e];   // contiguous per lane
  }
  return b;
}

// ---------------------------------------------------------------------------
// fp32 -> f16 conversion (grid-stride), plain and transposing variants
// ---------------------------------------------------------------------------
__global__ void cvt_f32_to_f16(const float* __restrict__ in,
                               _Float16* __restrict__ out, long n) {
  long i = (long)blockIdx.x * blockDim.x + threadIdx.x;
  const long stride = (long)gridDim.x * blockDim.x;
  for (; i < n; i += stride) out[i] = (_Float16)in[i];
}

// in: [K, N] row-major -> out: [N, K] (f16). Reads coalesced, writes strided.
__global__ void cvt_transpose_f16(const float* __restrict__ in,
                                  _Float16* __restrict__ out, int K, int N) {
  long i = (long)blockIdx.x * blockDim.x + threadIdx.x;
  const long total = (long)K * N;
  const long stride = (long)gridDim.x * blockDim.x;
  for (; i < total; i += stride) {
    const int k = (int)(i / N);
    const int n = (int)(i % N);
    out[(long)n * K + k] = (_Float16)in[i];
  }
}

// ---------------------------------------------------------------------------
// GEMM1: qkv = feats[8192,1024] @ Wqkv^T-stored + bqkv.
// 64x64 tile per wave; per K=32 step: 8 b128 (A) + 8 b128 (B), 16 WMMAs.
// __launch_bounds__(256,1): ~190 live VGPRs (128 acc) must NOT spill; lift
// the default occupancy cap (wave32 can address up to 1024 VGPRs).
// Epilogue: cols < 2048 (Q,K) -> row-major qk[N,2048];
//           cols >= 2048 (V)  -> TRANSPOSED vt[c][token] for the attention
//           V^T A-fragments. Branch is wave-uniform (n0 multiple of 64).
// ---------------------------------------------------------------------------
__global__ void __launch_bounds__(256, 1)
wmma_gemm_qkv(const _Float16* __restrict__ A,
              const _Float16* __restrict__ Bt,   // [3072][1024]
              const float* __restrict__ bias,
              _Float16* __restrict__ qk,         // [8192][2048]
              _Float16* __restrict__ vt) {       // [1024][8192]
  const int M = N_TOK, K = C_DIM, N = 3 * C_DIM;
  const int wave = threadIdx.x >> 5;
  const int lane = threadIdx.x & 31;
  const int ntiles = N >> 6;
  const long tile = (long)blockIdx.x * (blockDim.x >> 5) + wave;
  if (tile >= (long)(M >> 6) * ntiles) return;
  const int m0 = (int)(tile / ntiles) << 6;
  const int n0 = (int)(tile % ntiles) << 6;

  v8f acc[4][4] = {};
  for (int k0 = 0; k0 < K; k0 += 32) {
    v16h a[4];
#pragma unroll
    for (int mi = 0; mi < 4; ++mi)
      a[mi] = load_a_frag(A + (long)(m0 + (mi << 4)) * K + k0, K, lane);
    if (k0 + 32 < K)
      __builtin_prefetch((const void*)(Bt + (long)(n0 + lane) * K + k0 + 32), 0, 1);
#pragma unroll
    for (int t = 0; t < 4; ++t) {
      const v16h b = load_bt_frag(Bt + (long)(n0 + (t << 4)) * K + k0, K, lane);
#pragma unroll
      for (int mi = 0; mi < 4; ++mi)
        acc[mi][t] = wmma_f16(a[mi], b, acc[mi][t]);
    }
  }

  const int hb  = (lane >> 4) << 3;
  const int nlo = lane & 15;
#pragma unroll
  for (int t = 0; t < 4; ++t) {
    const int col = n0 + (t << 4) + nlo;
    const float bv = bias[col];
    if (n0 < 2 * C_DIM) {            // Q | K -> row-major
#pragma unroll
      for (int mi = 0; mi < 4; ++mi)
#pragma unroll
        for (int r = 0; r < 8; ++r) {
          const int row = m0 + (mi << 4) + r + hb;
          qk[(long)row * QK_LD + col] = (_Float16)(acc[mi][t][r] + bv);
        }
    } else {                         // V -> transposed [c][token]
#pragma unroll
      for (int mi = 0; mi < 4; ++mi)
#pragma unroll
        for (int r = 0; r < 8; ++r) {
          const int row = m0 + (mi << 4) + r + hb;
          vt[(long)(col - 2 * C_DIM) * VT_LD + row] = (_Float16)(acc[mi][t][r] + bv);
        }
    }
  }
}

// ---------------------------------------------------------------------------
// GEMM2: out = ctx[8192,1024] @ Wout^T-stored + bout -> f32
// ---------------------------------------------------------------------------
__global__ void __launch_bounds__(256, 1)
wmma_gemm_out(const _Float16* __restrict__ A,
              const _Float16* __restrict__ Bt,   // [1024][1024]
              const float* __restrict__ bias,
              float* __restrict__ Out) {
  const int M = N_TOK, K = C_DIM, N = C_DIM;
  const int wave = threadIdx.x >> 5;
  const int lane = threadIdx.x & 31;
  const int ntiles = N >> 6;
  const long tile = (long)blockIdx.x * (blockDim.x >> 5) + wave;
  if (tile >= (long)(M >> 6) * ntiles) return;
  const int m0 = (int)(tile / ntiles) << 6;
  const int n0 = (int)(tile % ntiles) << 6;

  v8f acc[4][4] = {};
  for (int k0 = 0; k0 < K; k0 += 32) {
    v16h a[4];
#pragma unroll
    for (int mi = 0; mi < 4; ++mi)
      a[mi] = load_a_frag(A + (long)(m0 + (mi << 4)) * K + k0, K, lane);
    if (k0 + 32 < K)
      __builtin_prefetch((const void*)(Bt + (long)(n0 + lane) * K + k0 + 32), 0, 1);
#pragma unroll
    for (int t = 0; t < 4; ++t) {
      const v16h b = load_bt_frag(Bt + (long)(n0 + (t << 4)) * K + k0, K, lane);
#pragma unroll
      for (int mi = 0; mi < 4; ++mi)
        acc[mi][t] = wmma_f16(a[mi], b, acc[mi][t]);
    }
  }

  const int hb  = (lane >> 4) << 3;
  const int nlo = lane & 15;
#pragma unroll
  for (int t = 0; t < 4; ++t) {
    const int col = n0 + (t << 4) + nlo;
    const float bv = bias[col];
#pragma unroll
    for (int mi = 0; mi < 4; ++mi)
#pragma unroll
      for (int r = 0; r < 8; ++r) {
        const int row = m0 + (mi << 4) + r + hb;
        Out[(long)row * N + col] = acc[mi][t][r] + bv;
      }
  }
}

// ---------------------------------------------------------------------------
// Flash attention, transposed-score formulation. One wave per (b, h, 16 q).
//   S^T = K @ (Q*scale)^T  -> softmax axis (kv) in-lane (+1 xor-16 each)
//   O^T = V^T @ P^T        -> alpha rescale and 1/l in-lane
// V comes pre-transposed (vt), so ALL fragment loads are 2x b128 per lane.
// P^T C-layout -> B-frag via 4 packed-f16 shfl_xor(16). No LDS memory use.
// ---------------------------------------------------------------------------
__global__ void __launch_bounds__(128, 1)
flash_attn_kernel(const _Float16* __restrict__ qk,
                  const _Float16* __restrict__ vt,
                  _Float16* __restrict__ ctx) {
  const int wave = threadIdx.x >> 5;
  const int lane = threadIdx.x & 31;
  const int tile = blockIdx.x * 4 + wave;        // 0 .. B*H*(S/16)-1, exact
  const int qt = tile & (S_SEQ / 16 - 1);
  const int h  = (tile >> 6) & (H_HEADS - 1);
  const int b  = tile >> 10;
  const int tok0 = b * S_SEQ + (qt << 4);

  const int q   = lane & 15;       // this lane's query column
  const int hi  = lane >> 4;
  const int khi = hi << 4;
  const int hb  = hi << 3;

  // ---- Q^T as two B-fragments, pre-scaled by 1/sqrt(64)=0.125 (exact) ----
  v16h bq0, bq1;
  {
    const _Float16* qrow = qk + (long)(tok0 + q) * QK_LD + h * D_HEAD;
#pragma unroll
    for (int e = 0; e < 16; ++e) {
      bq0[e] = (_Float16)((float)qrow[khi + e]      * 0.125f);
      bq1[e] = (_Float16)((float)qrow[32 + khi + e] * 0.125f);
    }
  }

  float mcol = -1.0e30f, lcol = 0.0f;            // per-lane (per-q) running stats
  v8f o[4] = {};                                 // O^T tiles: rows d, col q

  const _Float16* kb0 = qk + (long)(b * S_SEQ) * QK_LD + C_DIM + h * D_HEAD;
  const _Float16* vt0 = vt + (long)(h * D_HEAD) * VT_LD + b * S_SEQ;

  for (int kv0 = 0; kv0 < S_SEQ; kv0 += 32) {
    // ---- S^T tiles: s[nt] = K[kv tile nt] @ Qs^T  (4 WMMAs) ----
    v8f s[2] = {};
#pragma unroll
    for (int nt = 0; nt < 2; ++nt) {
      v16h a0, a1;   // A-frags of K tile: row m = kv-in-tile, K dim = d
      const _Float16* krow = kb0 + (long)(kv0 + (nt << 4) + q) * QK_LD;
#pragma unroll
      for (int e = 0; e < 16; ++e) {
        const int k = khi + e;
        a0[e] = krow[k];
        a1[e] = krow[32 + k];
      }
      s[nt] = wmma_f16(a0, bq0, s[nt]);
      s[nt] = wmma_f16(a1, bq1, s[nt]);
    }
    // lane holds 16 kv-scores of its q column: kv = 8hi+r (s0), 16+8hi+r (s1)

    // ---- online softmax: in-lane max/sum + one xor-16 each ----
    float mx = s[0][0];
#pragma unroll
    for (int r = 1; r < 8; ++r) mx = fmaxf(mx, s[0][r]);
#pragma unroll
    for (int r = 0; r < 8; ++r) mx = fmaxf(mx, s[1][r]);
    mx = fmaxf(mx, __shfl_xor(mx, 16, 32));
    const float mnew  = fmaxf(mcol, mx);
    const float alpha = __expf(mcol - mnew);

    float p0[8], p1[8], rs = 0.0f;
#pragma unroll
    for (int r = 0; r < 8; ++r) {
      p0[r] = __expf(s[0][r] - mnew);
      p1[r] = __expf(s[1][r] - mnew);
      rs += p0[r] + p1[r];
    }
    rs += __shfl_xor(rs, 16, 32);
    lcol = lcol * alpha + rs;
    mcol = mnew;

#pragma unroll
    for (int t = 0; t < 4; ++t)
#pragma unroll
      for (int r = 0; r < 8; ++r) o[t][r] *= alpha;   // in-lane: col is q

    // ---- build P^T B-frag: elem e -> kv = 16*hi + e ----
    union { unsigned u[8]; v16h v; } bp;
    unsigned ph0[4], ph1[4];
#pragma unroll
    for (int i = 0; i < 4; ++i) {
      union { _Float16 h[2]; unsigned u; } c0, c1;
      c0.h[0] = (_Float16)p0[2 * i]; c0.h[1] = (_Float16)p0[2 * i + 1];
      c1.h[0] = (_Float16)p1[2 * i]; c1.h[1] = (_Float16)p1[2 * i + 1];
      ph0[i] = c0.u; ph1[i] = c1.u;
    }
#pragma unroll
    for (int i = 0; i < 4; ++i) {
      const unsigned send = hi ? ph0[i] : ph1[i];     // what my partner needs
      const unsigned recv = (unsigned)__shfl_xor((int)send, 16, 32);
      bp.u[i]     = hi ? recv   : ph0[i];             // kv 16hi + 2i,2i+1
      bp.u[4 + i] = hi ? ph1[i] : recv;               // kv 16hi + 8+2i,...
    }

    // ---- O^T += V^T @ P^T : V^T A-frags are contiguous (2x b128) ----
#pragma unroll
    for (int t = 0; t < 4; ++t) {
      v16h av;    // A-frag of V^T tile t: row m = d_local, K dim = kv
      const _Float16* vrow = vt0 + (long)((t << 4) + q) * VT_LD + kv0;
#pragma unroll
      for (int e = 0; e < 16; ++e) {
        const int k = khi + e;
        av[e] = vrow[k];
      }
      o[t] = wmma_f16(av, bp.v, o[t]);
    }
  }

  // ---- normalize (in-lane) and store context f16, [N, C] layout ----
  const float inv = 1.0f / lcol;
  _Float16* crow = ctx + (long)(tok0 + q) * C_DIM + h * D_HEAD;
#pragma unroll
  for (int t = 0; t < 4; ++t)
#pragma unroll
    for (int r = 0; r < 8; ++r)
      crow[(t << 4) + r + hb] = (_Float16)(o[t][r] * inv);
}

// ---------------------------------------------------------------------------
// Launch: cvt(feats) + cvtT(Wqkv,Wout) -> GEMM1 -> attention -> GEMM2
// Workspace (f16): feats 16MB | WqkvT 6MB | WoutT 2MB | qk 32MB | vt 16MB
//                  | ctx 16MB   (~88MB total)
// ---------------------------------------------------------------------------
extern "C" void kernel_launch(void* const* d_in, const int* in_sizes, int n_in,
                              void* d_out, int out_size, void* d_ws, size_t ws_size,
                              hipStream_t stream) {
  const float* feats = (const float*)d_in[0];
  const float* Wqkv  = (const float*)d_in[1];
  const float* bqkv  = (const float*)d_in[2];
  const float* Wout  = (const float*)d_in[3];
  const float* bout  = (const float*)d_in[4];
  float* out = (float*)d_out;

  char* ws = (char*)d_ws;
  _Float16* feats_h = (_Float16*)ws;  ws += (long)N_TOK * C_DIM * 2;
  _Float16* wqkv_t  = (_Float16*)ws;  ws += (long)C_DIM * 3 * C_DIM * 2;
  _Float16* wout_t  = (_Float16*)ws;  ws += (long)C_DIM * C_DIM * 2;
  _Float16* qk_h    = (_Float16*)ws;  ws += (long)N_TOK * QK_LD * 2;
  _Float16* vt_h    = (_Float16*)ws;  ws += (long)C_DIM * VT_LD * 2;
  _Float16* ctx_h   = (_Float16*)ws;

  cvt_f32_to_f16<<<2048, 256, 0, stream>>>(feats, feats_h, (long)N_TOK * C_DIM);
  cvt_transpose_f16<<<2048, 256, 0, stream>>>(Wqkv, wqkv_t, C_DIM, 3 * C_DIM);
  cvt_transpose_f16<<<1024, 256, 0, stream>>>(Wout, wout_t, C_DIM, C_DIM);

  // GEMM1: qk (row-major) + vt (transposed V), fused bias
  {
    const long tiles = (long)(N_TOK / 64) * ((3 * C_DIM) / 64);  // 6144
    wmma_gemm_qkv<<<(int)(tiles / 8), 256, 0, stream>>>(
        feats_h, wqkv_t, bqkv, qk_h, vt_h);
  }

  // Flash attention: B*H*(S/16) = 8192 wave tiles, 4 waves per block
  flash_attn_kernel<<<(B_BATCH * H_HEADS * (S_SEQ / 16)) / 4, 128, 0, stream>>>(
      qk_h, vt_h, ctx_h);

  // GEMM2: out = ctx @ Wout + bout -> f32
  {
    const long tiles = (long)(N_TOK / 64) * (C_DIM / 64);        // 2048
    wmma_gemm_out<<<(int)(tiles / 8), 256, 0, stream>>>(
        ctx_h, wout_t, bout, out);
  }
}
